// LSTM_86320252715683
// MI455X (gfx1250) — matrix operand order repaired and checked by reading, not compiled, not run on hardware
//
#include <hip/hip_runtime.h>

typedef __attribute__((ext_vector_type(16))) _Float16 v16h;
typedef __attribute__((ext_vector_type(8)))  _Float16 v8h;
typedef __attribute__((ext_vector_type(8)))  float    v8f;

#define SEQ  2048
#define BATCH 128
#define IDIM  128
#define HID   128
#define G4    512
#define LDW   136   // padded LDS row stride (halfs): 272B -> 4-bank rotation/row, conflict-free b128

// ---------------------------------------------------------------------------
// Phase 1: xg = x @ V.T + (b + b2), stored as f16 in tile-major layout:
//   xg_ws[ (mtile*32 + ntile) * 256 + lane*8 .. +7 ]   (WMMA C/D fragment order)
// M = S*B = 262144 (16384 m-tiles), N = 512 (32 n-tiles), K = 128.
// Each wave computes a 16x64 strip (4 n-tiles) reusing its A fragment.
// ---------------------------------------------------------------------------
__global__ __launch_bounds__(256) void lstm_xg_gemm(
    const float* __restrict__ x,      // (S*B, 128) row-major
    const float* __restrict__ V,      // (512, 128) row-major
    const float* __restrict__ b1,
    const float* __restrict__ b2,
    _Float16* __restrict__ xg)        // tile-major f16 workspace
{
    const int gw   = (blockIdx.x * 256 + threadIdx.x) >> 5;  // global wave id
    const int lane = threadIdx.x & 31;
    const int mt   = gw >> 3;          // 0..16383
    const int ng   = gw & 7;           // 4 n-tiles each -> cols ng*64..ng*64+63
    const int m    = lane & 15;
    const int ksel = lane >> 4;

    const float* xrow = x + (long)(mt * 16 + m) * IDIM;

    v8f acc[4];
    #pragma unroll
    for (int t = 0; t < 4; ++t) {
        const int col = ng * 64 + t * 16 + (lane & 15);
        const float bv = b1[col] + b2[col];
        #pragma unroll
        for (int e = 0; e < 8; ++e) acc[t][e] = bv;
    }

    #pragma unroll
    for (int kc = 0; kc < 4; ++kc) {
        // A fragment (16x32 f16): lane = ksel*16+m holds K = k0..k0+7 and k0+16..k0+23
        const int k0 = kc * 32 + ksel * 8;
        v16h a;
        #pragma unroll
        for (int e = 0; e < 8; ++e) a[e]     = (_Float16)xrow[k0 + e];
        #pragma unroll
        for (int e = 0; e < 8; ++e) a[8 + e] = (_Float16)xrow[k0 + 16 + e];

        #pragma unroll
        for (int t = 0; t < 4; ++t) {
            // B fragment (32x16): lane holds col n, K = kb..kb+7 and kb+16..kb+23, B[k][n] = V[n][k]
            const int n  = ng * 64 + t * 16 + (lane & 15);
            const int kb = kc * 32 + ksel * 8;
            const float* vrow = V + (long)n * IDIM + kb;
            v16h bm;
            #pragma unroll
            for (int e = 0; e < 16; ++e) bm[e] = (_Float16)vrow[e + (e >= 8 ? 8 : 0)];
            acc[t] = __builtin_amdgcn_wmma_f32_16x16x32_f16(
                false, a, false, bm, (short)0, acc[t], false, false);
        }
    }

    #pragma unroll
    for (int t = 0; t < 4; ++t) {
        const int ntg = ng * 4 + t;
        v8h h16 = __builtin_convertvector(acc[t], v8h);
        *(v8h*)(xg + ((long)mt * 32 + ntg) * 256 + lane * 8) = h16;
    }
}

// ---------------------------------------------------------------------------
// Phase 2: persistent single-workgroup recurrent scan.
// 32 waves, each owns 2 cells of the 8x8 (batch-tile x hidden-tile) grid.
// W (f16, padded) and h (f16, padded) live in LDS; c lives in VGPRs.
// Per step/wave: 2 cells * 4 gates * 4 k-chunks = 32 v_wmma_f32_16x16x32_f16.
// ---------------------------------------------------------------------------
__global__ __launch_bounds__(1024) void lstm_scan(
    const _Float16* __restrict__ xg,  // tile-major (phase-1 output)
    const float* __restrict__ W,      // (512, 128) row-major
    float* __restrict__ out)          // h_seq (S*B*H) | h_T (B*H) | c_T (B*H)
{
    __shared__ _Float16 Wlds[G4 * LDW];    // 512 rows * 272B = 139264 B
    __shared__ _Float16 hlds[HID * LDW];   // 128 rows * 272B =  34816 B

    const int tid  = threadIdx.x;
    const int wave = tid >> 5;
    const int lane = tid & 31;
    const int mfrag = lane & 15;
    const int ksel  = lane >> 4;

    // Stage W into LDS as f16 (one-time)
    for (int i = tid; i < G4 * HID; i += 1024) {
        const int n = i >> 7, k = i & 127;
        Wlds[n * LDW + k] = (_Float16)W[i];
    }
    // h0 = 0
    for (int i = tid; i < HID * LDW; i += 1024) hlds[i] = (_Float16)0.f;
    __syncthreads();

    v8f cstate[2];
    #pragma unroll
    for (int cc = 0; cc < 2; ++cc)
        #pragma unroll
        for (int e = 0; e < 8; ++e) cstate[cc][e] = 0.f;

    const long hT_off = (long)SEQ * BATCH * HID;
    const long cT_off = hT_off + (long)BATCH * HID;

    for (int s = 0; s < SEQ; ++s) {
        v8h hnew[2];
        #pragma unroll
        for (int cc = 0; cc < 2; ++cc) {
            const int cell = wave * 2 + cc;
            const int bt = cell >> 3;      // batch tile 0..7
            const int nt = cell & 7;       // hidden tile 0..7

            // Init gate accumulators from xg tiles (f16, fragment-ordered)
            v8f acc[4];
            #pragma unroll
            for (int g = 0; g < 4; ++g) {
                const _Float16* cp =
                    xg + (((long)(s * 8 + bt) * 32) + g * 8 + nt) * 256 + lane * 8;
                v8h c16 = *(const v8h*)cp;
                acc[g] = __builtin_convertvector(c16, v8f);
            }
            if (s + 1 < SEQ) {
                // warm L2/WGP$ for the next step's xg tiles
                __builtin_prefetch(
                    xg + (((long)((s + 1) * 8 + bt) * 32) + nt) * 256 + lane * 8, 0, 0);
            }

            // gates += h @ W.T   (K = 128, 4 chunks of 32)
            #pragma unroll
            for (int kc = 0; kc < 4; ++kc) {
                const int k0 = kc * 32 + ksel * 8;
                const _Float16* hrow = hlds + (bt * 16 + mfrag) * LDW;
                v8h alo = *(const v8h*)(hrow + k0);
                v8h ahi = *(const v8h*)(hrow + k0 + 16);
                v16h a = __builtin_shufflevector(alo, ahi,
                    0, 1, 2, 3, 4, 5, 6, 7, 8, 9, 10, 11, 12, 13, 14, 15);

                #pragma unroll
                for (int g = 0; g < 4; ++g) {
                    const int n  = g * HID + nt * 16 + (lane & 15); // row of W
                const int kb = kc * 32 + ksel * 8;
                    const _Float16* wrow = Wlds + n * LDW + kb;
                    v8h blo = *(const v8h*)wrow;
                v8h bhi = *(const v8h*)(wrow + 16);
                    v16h bm = __builtin_shufflevector(blo, bhi,
                        0, 1, 2, 3, 4, 5, 6, 7, 8, 9, 10, 11, 12, 13, 14, 15);
                    acc[g] = __builtin_amdgcn_wmma_f32_16x16x32_f16(
                        false, a, false, bm, (short)0, acc[g], false, false);
                }
            }

            // LSTM cell math on this wave's 8 elements per lane
            v8f cs = cstate[cc];
            v8f hf;
            #pragma unroll
            for (int e = 0; e < 8; ++e) {
                const float iv = 1.f / (1.f + __expf(-acc[0][e]));
                const float fv = 1.f / (1.f + __expf(-acc[1][e]));
                const float gv = tanhf(acc[2][e]);
                const float ov = 1.f / (1.f + __expf(-acc[3][e]));
                const float cn = fv * cs[e] + iv * gv;
                cs[e] = cn;
                hf[e] = ov * tanhf(cn);
            }
            cstate[cc] = cs;
            hnew[cc] = __builtin_convertvector(hf, v8h);

            // h_seq output (f32)
            {
                const int m0 = ksel * 8;
                const int n  = nt * 16 + (lane & 15);
                float* orow = out + ((long)s * BATCH + bt * 16 + m0) * HID + n;
                #pragma unroll
                for (int e = 0; e < 8; ++e) orow[(long)e * HID] = hf[e];
            }
            if (s == SEQ - 1) {
                const int m0 = ksel * 8;
                const int n  = nt * 16 + (lane & 15);
                #pragma unroll
                for (int e = 0; e < 8; ++e) {
                    const int bidx = bt * 16 + m0 + e;
                    out[hT_off + (long)bidx * HID + n] = hf[e];
                    out[cT_off + (long)bidx * HID + n] = cs[e];
                }
            }
        }

        __syncthreads();   // all waves done reading h(t-1) from LDS
        #pragma unroll
        for (int cc = 0; cc < 2; ++cc) {
            const int cell = wave * 2 + cc;
            const int bt = cell >> 3, nt = cell & 7;
            const int m0 = ksel * 8;
            const int n  = nt * 16 + (lane & 15);
            #pragma unroll
            for (int e = 0; e < 8; ++e)
                hlds[(bt * 16 + m0 + e) * LDW + n] = hnew[cc][e];
        }
        __syncthreads();   // h(t) visible to all waves
    }
}

extern "C" void kernel_launch(void* const* d_in, const int* in_sizes, int n_in,
                              void* d_out, int out_size, void* d_ws, size_t ws_size,
                              hipStream_t stream) {
    const float* x  = (const float*)d_in[0];   // (2048, 128, 128)
    const float* V  = (const float*)d_in[1];   // (512, 128)
    const float* W  = (const float*)d_in[2];   // (512, 128)
    const float* b  = (const float*)d_in[3];   // (512,)
    const float* b2 = (const float*)d_in[4];   // (512,)
    float* out = (float*)d_out;
    _Float16* xg = (_Float16*)d_ws;            // 2048*8*32*256 f16 = 256 MiB

    // Phase 1: 131072 waves = 16384 blocks x 8 waves
    lstm_xg_gemm<<<16384, 256, 0, stream>>>(x, V, b, b2, xg);
    // Phase 2: one persistent workgroup (32 waves on one WGP)
    lstm_scan<<<1, 1024, 0, stream>>>(xg, W, out);
}
